// SSM_2645699854376
// MI455X (gfx1250) — matrix operand order repaired and checked by reading, not compiled, not run on hardware
//
#include <hip/hip_runtime.h>
#include <hip/hip_bf16.h>

// ---------------------------------------------------------------------------
// Bidirectional Mamba-3D block for MI455X (gfx1250, wave32, WMMA).
// GEMMs: v_wmma_f32_16x16x32_f16, f16 operands pre-converted once, f32 acc.
// Each wave computes a 16x(NT*16) tile: A-fragment reused across NT WMMAs.
// Weight tiles DMA'd into LDS with GLOBAL_LOAD_ASYNC_TO_LDS_B128 (ASYNCcnt).
// ---------------------------------------------------------------------------

typedef __attribute__((ext_vector_type(16))) _Float16 v16h;
typedef __attribute__((ext_vector_type(8)))  float    v8f;
typedef _Float16 h1;

namespace {
constexpr int kDim    = 64;
constexpr int kDInner = 128;
constexpr int kDiM    = 256;   // 2*D_INNER
constexpr int kB      = 4;
constexpr int kT      = 4;
constexpr int kH      = 32;
constexpr int kW      = 16;
constexpr int kL      = kT * kH * kW;  // 2048
constexpr int kB2     = 2 * kB;        // 8 (fwd + rev)
constexpr int kBL     = kB * kL;       // 8192
constexpr int kB2L    = kB2 * kL;      // 16384
constexpr int kXK     = 40;            // DT_RANK + 2*D_STATE
constexpr int kXKp    = 48;            // padded to 3 N-tiles
}

__device__ __forceinline__ float siluf_(float x) {
  // x * sigmoid(x) with v_rcp_f32 instead of precise IEEE divide.
  return x * __builtin_amdgcn_rcpf(1.0f + __expf(-x));
}
__device__ __forceinline__ float softplusf_(float x) {
  return (x > 20.f) ? x : __logf(1.0f + __expf(x));
}

// ---------------------------------------------------------------------------
// WMMA fragment layouts (ISA 7.12.2, wave32):
//  A (16x32 f16): lanes 0-15 -> M=lane; VGPR j<4 -> K=2j,2j+1 (+8 upper lanes),
//                 VGPR j>=4 -> K=2j+8 (+8 upper lanes).
//  B (32x16 f16): N=lane&15; same K mapping as A (VGPR v<4 -> K=2v, v>=4 -> K=2v+8; +8 upper lanes).
//  C/D (16x16 f32): VGPR r -> M=r (lanes 0-15) / r+8 (lanes 16-31), N=lane&15.
//
// gemm_core: block DMAs the NT*16 x K f16 weight tile into LDS via the async
// pipe (each wave drains its own ASYNCcnt before the barrier); each of the 8
// waves computes one 16-row M tile against all NT N-tiles (A reuse).
// ---------------------------------------------------------------------------
template <int K, int NT, typename FA, typename ST>
__device__ __forceinline__ void gemm_core(h1* lds, const h1* __restrict__ w16,
                                          int n0g, int m0, FA&& fetchA, ST&& store) {
  {
    // Contiguous NT*16*K*2 bytes (ldw == K); multiples of 4096 per 256-thread
    // sweep, so the loop is EXEC-uniform.  ISA: LDS aperture address = addr[31:0].
    const int bytes = NT * 16 * K * (int)sizeof(h1);
    const unsigned int lds0 = (unsigned int)(uintptr_t)lds;
    const unsigned long long g0 =
        (unsigned long long)(uintptr_t)(w16 + (size_t)n0g * K);
    for (int off = (int)threadIdx.x * 16; off < bytes; off += 256 * 16) {
      unsigned int       ldsoff = lds0 + (unsigned int)off;
      unsigned long long gaddr  = g0 + (unsigned long long)off;
      asm volatile("global_load_async_to_lds_b128 %0, %1, off"
                   :: "v"(ldsoff), "v"(gaddr)
                   : "memory");
    }
    asm volatile("s_wait_asynccnt 0x0" ::: "memory");
  }
  __syncthreads();

  const int lane = threadIdx.x & 31;
  const int half = lane >> 4;
  const int idx  = lane & 15;

  v8f acc[NT];
#pragma unroll
  for (int t = 0; t < NT; ++t) acc[t] = v8f{};

#pragma unroll
  for (int kc = 0; kc < K; kc += 32) {
    v16h a{};
#pragma unroll
    for (int j = 0; j < 8; ++j) {
      const int ka = kc + ((j < 4) ? (2 * j) : (2 * j + 8)) + 8 * half;
      a[2 * j]     = fetchA(idx, ka);
      a[2 * j + 1] = fetchA(idx, ka + 1);
    }
#pragma unroll
    for (int t = 0; t < NT; ++t) {
      v16h b{};
#pragma unroll
      for (int v = 0; v < 8; ++v) {
        const int kb = kc + ((v < 4) ? (2 * v) : (2 * v + 8)) + 8 * half;
        b[2 * v]     = lds[(t * 16 + idx) * K + kb];
        b[2 * v + 1] = lds[(t * 16 + idx) * K + kb + 1];
      }
      acc[t] = __builtin_amdgcn_wmma_f32_16x16x32_f16(false, a, false, b, (short)0,
                                                      acc[t], false, false);
    }
  }
#pragma unroll
  for (int t = 0; t < NT; ++t)
#pragma unroll
    for (int r = 0; r < 8; ++r)
      store(m0 + r + 8 * half, n0g + t * 16 + idx, acc[t][r]);
}

// ---------------------------------------------------------------------------
// f32 -> f16 conversion kernels (weights + x, run once at the head).
// ---------------------------------------------------------------------------
__global__ void k_cvt(const float* __restrict__ s, h1* __restrict__ d, int n) {
  const int i = blockIdx.x * blockDim.x + threadIdx.x;
  if (i < n) d[i] = (h1)s[i];
}
// x_proj_w padded from 40 to 48 rows (zero rows -> zero output columns).
__global__ void k_cvt_pad(const float* __restrict__ s, h1* __restrict__ d) {
  const int i = blockIdx.x * blockDim.x + threadIdx.x;  // 48*256 = 12288
  const int r = i >> 8;
  d[i] = (r < kXK) ? (h1)s[i] : (h1)0.0f;
}

// ---------------------------------------------------------------------------
// 1) xz = x @ in_proj_w.T ; xc = xz[:,:128] (f32) ; z1 = silu(xz[:,128:]).
//    M=8192, N=256 (4 macros of 64), K=64.  Split is macro-uniform.
// ---------------------------------------------------------------------------
__global__ void k_inproj1(const h1* __restrict__ x16, const h1* __restrict__ w16,
                          float* __restrict__ xc, float* __restrict__ z1) {
  __shared__ h1 lds[4 * 16 * kDim];
  const int nmac = blockIdx.x & 3;
  const int mg   = blockIdx.x >> 2;
  const int m0   = (mg * 8 + (threadIdx.x >> 5)) * 16;
  const int n0g  = nmac * 64;
  if (nmac < 2) {
    gemm_core<kDim, 4>(lds, w16, n0g, m0,
        [&](int i, int k) { return x16[(size_t)(m0 + i) * kDim + k]; },
        [&](int m, int n, float v) { xc[(size_t)m * kDInner + n] = v; });
  } else {
    gemm_core<kDim, 4>(lds, w16, n0g, m0,
        [&](int i, int k) { return x16[(size_t)(m0 + i) * kDim + k]; },
        [&](int m, int n, float v) {
          z1[(size_t)m * kDInner + (n - kDInner)] = siluf_(v);
        });
  }
}

// ---------------------------------------------------------------------------
// 2) depthwise conv3d 3x3x3 pad 1 + bias + silu -> seq16 (f16, layout (b,l,c)).
// ---------------------------------------------------------------------------
__global__ void k_conv3d(const float* __restrict__ xc, const float* __restrict__ cw,
                         const float* __restrict__ cb, h1* __restrict__ seq16) {
  const int gid = blockIdx.x * blockDim.x + threadIdx.x;   // 4*2048*128
  const int c = gid & 127;
  const int l = (gid >> 7) & 2047;
  const int b = gid >> 18;
  const int w = l & 15, h = (l >> 4) & 31, t = l >> 9;
  float s = cb[c];
#pragma unroll
  for (int dt = -1; dt <= 1; ++dt) {
    const int tt = t + dt;
    if (tt < 0 || tt >= kT) continue;
#pragma unroll
    for (int dh = -1; dh <= 1; ++dh) {
      const int hh = h + dh;
      if (hh < 0 || hh >= kH) continue;
#pragma unroll
      for (int dw = -1; dw <= 1; ++dw) {
        const int ww = w + dw;
        if (ww < 0 || ww >= kW) continue;
        const int lp = (tt * kH + hh) * kW + ww;
        s += cw[c * 27 + (dt + 1) * 9 + (dh + 1) * 3 + (dw + 1)] *
             xc[((size_t)b * kL + lp) * kDInner + c];
      }
    }
  }
  seq16[gid] = (h1)siluf_(s);
}

// ---------------------------------------------------------------------------
// 3) mamba in_proj: (fwd|rev seq16) @ m_in_proj_w.T -> u_pre / z2 (f32, (b2,d,l)).
//    M=16384, N=512 (8 macros), K=128.  Split at 256 is macro-uniform.
// ---------------------------------------------------------------------------
__global__ void k_minproj(const h1* __restrict__ seq16, const h1* __restrict__ w16,
                          float* __restrict__ u_pre, float* __restrict__ z2) {
  __shared__ h1 lds[4 * 16 * kDInner];
  const int nmac = blockIdx.x & 7;
  const int mg   = blockIdx.x >> 3;
  const int m0   = (mg * 8 + (threadIdx.x >> 5)) * 16;
  const int n0g  = nmac * 64;
  auto fa = [&](int i, int k) {
    const int m = m0 + i;
    const int b2 = m >> 11, l = m & 2047;
    const int lb = (b2 >= kB) ? (kL - 1 - l) : l;
    const int b = b2 & (kB - 1);
    return seq16[((size_t)b * kL + lb) * kDInner + k];
  };
  if (nmac < 4) {
    gemm_core<kDInner, 4>(lds, w16, n0g, m0, fa, [&](int m, int n, float v) {
      const int b2 = m >> 11, l = m & 2047;
      u_pre[((size_t)b2 * kDiM + n) * kL + l] = v;
    });
  } else {
    gemm_core<kDInner, 4>(lds, w16, n0g, m0, fa, [&](int m, int n, float v) {
      const int b2 = m >> 11, l = m & 2047;
      z2[((size_t)b2 * kDiM + (n - kDiM)) * kL + l] = v;
    });
  }
}

// ---------------------------------------------------------------------------
// 4) causal depthwise conv1d (k=4) + bias + silu -> u (f32) and u16 (f16).
// ---------------------------------------------------------------------------
__global__ void k_conv1d(const float* __restrict__ up, const float* __restrict__ cw,
                         const float* __restrict__ cb, float* __restrict__ u,
                         h1* __restrict__ u16) {
  const int gid = blockIdx.x * blockDim.x + threadIdx.x;   // 8*256*2048
  const int l = gid & 2047;
  const int d = (gid >> 11) & 255;
  float s = cb[d];
#pragma unroll
  for (int j = 0; j < 4; ++j) {
    const int ls = l - 3 + j;
    if (ls >= 0) s += cw[d * 4 + j] * up[gid - l + ls];
  }
  s = siluf_(s);
  u[gid] = s;
  u16[gid] = (h1)s;
}

// ---------------------------------------------------------------------------
// 5) x_dbl = x_proj_w @ u  (as out^T GEMM).  M=16384, N=48(pad40), K=256.
// ---------------------------------------------------------------------------
__global__ void k_xproj(const h1* __restrict__ u16, const h1* __restrict__ xw16,
                        float* __restrict__ xdbl) {
  __shared__ h1 lds[kXKp * kDiM];
  const int m0 = (blockIdx.x * 8 + (threadIdx.x >> 5)) * 16;
  gemm_core<kDiM, 3>(lds, xw16, 0, m0,
      [&](int i, int k) {
        const int m = m0 + i;
        const int b2 = m >> 11, l = m & 2047;
        return u16[((size_t)b2 * kDiM + k) * kL + l];
      },
      [&](int m, int n, float v) {
        if (n < kXK) {
          const int b2 = m >> 11, l = m & 2047;
          xdbl[((size_t)b2 * kXK + n) * kL + l] = v;
        }
      });
}

// ---------------------------------------------------------------------------
// 6) delta = softplus(dt_proj_w @ dt + b).  K=8 -> scalar.  (b2,d,l).
// ---------------------------------------------------------------------------
__global__ void k_delta(const float* __restrict__ xdbl, const float* __restrict__ dtw,
                        const float* __restrict__ dtb, float* __restrict__ delta) {
  const int gid = blockIdx.x * blockDim.x + threadIdx.x;
  const int l = gid & 2047;
  const int d = (gid >> 11) & 255;
  const int b2 = gid >> 19;
  float s = dtb[d];
#pragma unroll
  for (int r = 0; r < 8; ++r)
    s += dtw[d * 8 + r] * xdbl[((size_t)b2 * kXK + r) * kL + l];
  delta[gid] = softplusf_(s);
}

// ---------------------------------------------------------------------------
// 7) selective scan.  One state (n) per lane; half-wave = one (b2,d) row.
//    h = exp(delta*A)*h + delta*B*u ;  y = sum_n h*C + D*u  (y overwrites u).
// ---------------------------------------------------------------------------
__global__ void k_scan(const float* __restrict__ delta, float* __restrict__ u,
                       const float* __restrict__ xdbl, const float* __restrict__ alog,
                       const float* __restrict__ dvec) {
  const int wave = (blockIdx.x * blockDim.x + threadIdx.x) >> 5;  // 0..1023
  const int lane = threadIdx.x & 31;
  const int half = lane >> 4;
  const int n    = lane & 15;
  const int pair = wave * 2 + half;
  const int b2 = pair >> 8, d = pair & 255;

  const float a  = -__expf(alog[d * 16 + n]);
  const float Dd = dvec[d];
  const float* dl = delta + ((size_t)b2 * kDiM + d) * kL;
  float*       ul = u     + ((size_t)b2 * kDiM + d) * kL;
  const float* Bl = xdbl  + ((size_t)b2 * kXK + 8 + n) * kL;
  const float* Cl = xdbl  + ((size_t)b2 * kXK + 24 + n) * kL;

  float h = 0.0f;
  for (int l = 0; l < kL; ++l) {
    if ((l & 63) == 0 && l + 64 < kL) {
      __builtin_prefetch(dl + l + 64, 0, 0);
      __builtin_prefetch(ul + l + 64, 0, 0);
      __builtin_prefetch(Bl + l + 64, 0, 0);
      __builtin_prefetch(Cl + l + 64, 0, 0);
    }
    const float dv = dl[l];
    const float uv = ul[l];
    h = __expf(dv * a) * h + dv * Bl[l] * uv;
    float c = h * Cl[l];
    c += __shfl_xor(c, 1, 32);
    c += __shfl_xor(c, 2, 32);
    c += __shfl_xor(c, 4, 32);
    c += __shfl_xor(c, 8, 32);
    if (n == 0) ul[l] = c + Dd * uv;          // y in place over u
  }
}

// ---------------------------------------------------------------------------
// 8) gating: g16 = (f16)(y * silu(z2))   (y lives in u).
// ---------------------------------------------------------------------------
__global__ void k_gate(const float* __restrict__ y, const float* __restrict__ z2,
                       h1* __restrict__ g16) {
  const int gid = blockIdx.x * blockDim.x + threadIdx.x;
  g16[gid] = (h1)(y[gid] * siluf_(z2[gid]));
}

// ---------------------------------------------------------------------------
// 9) mamba out_proj: mixed = g @ m_out_proj_w.T.  M=16384, N=128 (2 macros), K=256.
// ---------------------------------------------------------------------------
__global__ void k_moutproj(const h1* __restrict__ g16, const h1* __restrict__ w16,
                           float* __restrict__ mixed) {
  __shared__ h1 lds[4 * 16 * kDiM];
  const int nmac = blockIdx.x & 1;
  const int mg   = blockIdx.x >> 1;
  const int m0   = (mg * 8 + (threadIdx.x >> 5)) * 16;
  gemm_core<kDiM, 4>(lds, w16, nmac * 64, m0,
      [&](int i, int k) {
        const int m = m0 + i;
        const int b2 = m >> 11, l = m & 2047;
        return g16[((size_t)b2 * kDiM + k) * kL + l];
      },
      [&](int m, int n, float v) { mixed[(size_t)m * kDInner + n] = v; });
}

// ---------------------------------------------------------------------------
// 10) combine fwd + reversed rev, multiply by silu(z1) -> ycomb16 (f16).
// ---------------------------------------------------------------------------
__global__ void k_combine(const float* __restrict__ mixed, const float* __restrict__ z1,
                          h1* __restrict__ yc16) {
  const int gid = blockIdx.x * blockDim.x + threadIdx.x;   // 4*2048*128
  const int c = gid & 127;
  const int l = (gid >> 7) & 2047;
  const int b = gid >> 18;
  const float fwd = mixed[((size_t)b * kL + l) * kDInner + c];
  const float rev = mixed[((size_t)(b + kB) * kL + (kL - 1 - l)) * kDInner + c];
  yc16[gid] = (h1)((fwd + rev) * z1[gid]);
}

// ---------------------------------------------------------------------------
// 11) out = ycomb @ out_proj_w.T.  M=8192, N=64 (1 macro), K=128.
// ---------------------------------------------------------------------------
__global__ void k_outproj(const h1* __restrict__ yc16, const h1* __restrict__ w16,
                          float* __restrict__ out) {
  __shared__ h1 lds[4 * 16 * kDInner];
  const int m0 = (blockIdx.x * 8 + (threadIdx.x >> 5)) * 16;
  gemm_core<kDInner, 4>(lds, w16, 0, m0,
      [&](int i, int k) { return yc16[(size_t)(m0 + i) * kDInner + k]; },
      [&](int m, int n, float v) { out[(size_t)m * kDim + n] = v; });
}

// ---------------------------------------------------------------------------
extern "C" void kernel_launch(void* const* d_in, const int* in_sizes, int n_in,
                              void* d_out, int out_size, void* d_ws, size_t ws_size,
                              hipStream_t stream) {
  const float* x      = (const float*)d_in[0];
  const float* inw    = (const float*)d_in[1];
  const float* c3w    = (const float*)d_in[2];
  const float* c3b    = (const float*)d_in[3];
  const float* outw   = (const float*)d_in[4];
  const float* minw   = (const float*)d_in[5];
  const float* c1w    = (const float*)d_in[6];
  const float* c1b    = (const float*)d_in[7];
  const float* xprojw = (const float*)d_in[8];
  const float* dtw    = (const float*)d_in[9];
  const float* dtb    = (const float*)d_in[10];
  const float* alog   = (const float*)d_in[11];
  const float* dvec   = (const float*)d_in[12];
  const float* moutw  = (const float*)d_in[13];
  float* out = (float*)d_out;

  // --- workspace layout: f32 region first, then f16 region -----------------
  float* ws    = (float*)d_ws;
  float* xc    = ws;                              // 8192*128
  float* z1    = xc   + (size_t)kBL * kDInner;    // 8192*128
  float* upre  = z1   + (size_t)kBL * kDInner;    // 8*256*2048 (-> delta)
  float* z2    = upre + (size_t)kB2 * kDiM * kL;  // 8*256*2048
  float* u     = z2   + (size_t)kB2 * kDiM * kL;  // 8*256*2048 (-> y in place)
  float* xdbl  = u    + (size_t)kB2 * kDiM * kL;  // 8*40*2048
  float* mixed = xdbl + (size_t)kB2 * kXK * kL;   // 16384*128
  h1* h16base  = (h1*)(mixed + (size_t)kB2L * kDInner);
  h1* x16      = h16base;                         // 8192*64
  h1* inw16    = x16   + (size_t)kBL * kDim;      // 256*64
  h1* minw16   = inw16 + 256 * kDim;              // 512*128
  h1* xw16     = minw16 + 512 * kDInner;          // 48*256 (padded)
  h1* moutw16  = xw16  + kXKp * kDiM;             // 128*256
  h1* outw16   = moutw16 + kDInner * kDiM;        // 64*128
  h1* seq16    = outw16 + kDim * kDInner;         // 8192*128
  h1* u16      = seq16 + (size_t)kBL * kDInner;   // 8*256*2048
  h1* g16      = u16   + (size_t)kB2 * kDiM * kL; // 8*256*2048
  h1* yc16     = g16   + (size_t)kB2 * kDiM * kL; // 8192*128

  // --- one-time f32 -> f16 conversions -------------------------------------
  k_cvt    <<<2048, 256, 0, stream>>>(x,    x16,    kBL * kDim);
  k_cvt    <<<64,   256, 0, stream>>>(inw,  inw16,  256 * kDim);
  k_cvt    <<<256,  256, 0, stream>>>(minw, minw16, 512 * kDInner);
  k_cvt_pad<<<48,   256, 0, stream>>>(xprojw, xw16);
  k_cvt    <<<128,  256, 0, stream>>>(moutw, moutw16, kDInner * kDiM);
  k_cvt    <<<32,   256, 0, stream>>>(outw, outw16, kDim * kDInner);

  // --- pipeline -------------------------------------------------------------
  k_inproj1 <<<256,   256, 0, stream>>>(x16, inw16, xc, z1);
  k_conv3d  <<<4096,  256, 0, stream>>>(xc, c3w, c3b, seq16);
  k_minproj <<<1024,  256, 0, stream>>>(seq16, minw16, upre, z2);
  k_conv1d  <<<16384, 256, 0, stream>>>(upre, c1w, c1b, u, u16);
  k_xproj   <<<128,   256, 0, stream>>>(u16, xw16, xdbl);
  k_delta   <<<16384, 256, 0, stream>>>(xdbl, dtw, dtb, upre /* delta */);
  k_scan    <<<128,   256, 0, stream>>>(upre, u, xdbl, alog, dvec);
  k_gate    <<<16384, 256, 0, stream>>>(u, z2, g16);
  k_moutproj<<<256,   256, 0, stream>>>(g16, moutw16, mixed);
  k_combine <<<4096,  256, 0, stream>>>(mixed, z1, yc16);
  k_outproj <<<64,    256, 0, stream>>>(yc16, outw16, out);
}
